// RNNAttention_46059229282556
// MI455X (gfx1250) — compile-verified, matrix operand order checked by reading, not run on hardware
//
#include <hip/hip_runtime.h>
#include <hip/hip_bf16.h>
#include <math.h>
#include <stdint.h>

typedef __bf16 bf16_t;
typedef __attribute__((ext_vector_type(16))) __bf16 v16bf;
typedef __attribute__((ext_vector_type(8)))  __bf16 v8bf;
typedef __attribute__((ext_vector_type(8)))  float  v8f;
typedef __attribute__((ext_vector_type(4)))  unsigned int u32x4;
typedef __attribute__((ext_vector_type(8)))  int i32x8;
typedef __attribute__((ext_vector_type(4)))  int i32x4;

#define BB 4
#define SS 1024
#define EE 1024
#define HH 1024
#define VV 32000

// GEMM block tile
#define BM 64      // 4 M-waves * 16
#define BN 128     // 2 N-wave-groups * 64
#define BK 64      // K slice staged in LDS (two WMMA K-steps)

#if __has_builtin(__builtin_amdgcn_tensor_load_to_lds) && __has_builtin(__builtin_amdgcn_s_wait_tensorcnt)
#define USE_TDM 1
#else
#define USE_TDM 0
#endif

// ---------------- elementwise f32 -> bf16 ----------------
__global__ void k_f32_to_bf16(const float* __restrict__ src, bf16_t* __restrict__ dst, long long n) {
  long long i = (long long)blockIdx.x * blockDim.x + threadIdx.x;
  if (i < n) dst[i] = (bf16_t)src[i];
}

// ---------------- embedding gather + bf16 cast ----------------
__global__ void k_embed(const int* __restrict__ tokens, const float* __restrict__ emb,
                        bf16_t* __restrict__ xb, long long n) {
  long long i = (long long)blockIdx.x * blockDim.x + threadIdx.x; // over B*S*E
  if (i >= n) return;
  int row = (int)(i >> 10);      // E == 1024
  int e   = (int)(i & 1023);
  int tok = tokens[row];
  xb[i] = (bf16_t)emb[(long long)tok * EE + e];
}

// ---------------- zero GRU state ----------------
__global__ void k_init_state(float* __restrict__ h, bf16_t* __restrict__ h_b) {
  int i = blockIdx.x * blockDim.x + threadIdx.x;
  if (i < BB * HH) h[i] = 0.0f;
  if (i < 16 * HH) h_b[i] = (bf16_t)0.0f;   // padded to M=16 rows for WMMA
}

// ---------------- TDM issue: 2D tile of B into LDS ----------------
// Loads a BN x BK bf16 tile (viewed as 4-byte units: tile 32 x 128) to LDS.
__device__ __forceinline__ void tdm_issue_tile(const bf16_t* gtile, unsigned ldsOff, int ldb) {
#if USE_TDM
  unsigned long long ga = (unsigned long long)(uintptr_t)gtile;      // byte address
  u32x4 g0 = { 1u,                                                   // count=1, user mode
               ldsOff,                                               // lds_addr (bytes)
               (unsigned)(ga & 0xffffffffu),
               ((unsigned)(ga >> 32) & 0x01ffffffu) | 0x80000000u }; // addr hi | type=2
  i32x8 g1 = { 0x20000,                 // data_size = 2 (4 bytes), mask=0, no flags
               (BK / 2) << 16,          // tensor_dim0[15:0]  (4B units per row we touch)
               BN << 16,                // tensor_dim0 hi=0 | tensor_dim1[15:0]
               (BK / 2) << 16,          // tensor_dim1 hi=0 | tile_dim0
               BN,                      // tile_dim1 | tile_dim2=0
               ldb >> 1,                // tensor_dim0_stride lo (row stride in 4B units)
               0,                       // stride0 hi | stride1 lo
               0 };
  i32x4 z4 = {};
#if __clang_major__ >= 23
  i32x8 z8 = {};
  __builtin_amdgcn_tensor_load_to_lds(g0, g1, z4, z4, z8, 0);
#else
  __builtin_amdgcn_tensor_load_to_lds(g0, g1, z4, z4, 0);
#endif
#else
  (void)gtile; (void)ldsOff; (void)ldb;
#endif
}

// ---------------- generic batched WMMA GEMM: C = act((A * B^T + bias) * scale) ----------------
// A: [M,K] bf16 row-major (lda), B: [N,K] bf16 row-major (ldb)
// Cf: optional f32 out (ldcf), Cb: optional bf16 out (ldcb). act: 0=none, 1=tanh.
// Block: 256 threads = 8 waves, 4(M) x 2(N-group); block tile 64 x 128.
// B tile (BN x BK) staged in LDS via TDM double-buffer; A fragments direct from global.
__global__ __launch_bounds__(256) void k_wmma_gemm_bt(
    const bf16_t* __restrict__ A, const bf16_t* __restrict__ B,
    const float* __restrict__ bias,
    float* __restrict__ Cf, bf16_t* __restrict__ Cb,
    int M, int N, int K, int lda, int ldb, int ldcf, int ldcb,
    long long strideA, long long strideB,
    long long strideCf, long long strideCb,
    float scale, int act) {
  __shared__ __align__(16) bf16_t sB[2][BN * BK];   // 2 x 16 KB

  const int bId = blockIdx.z;
  A += (long long)bId * strideA;
  B += (long long)bId * strideB;

  const int tid   = threadIdx.x;
  const int wave  = tid >> 5;
  const int lane  = tid & 31;
  const int waveM = wave & 3;        // 0..3
  const int waveNg = wave >> 2;      // 0..1  (each group = 4 x 16 cols)
  const int rowBase  = blockIdx.y * BM + waveM * 16;
  const int colBlock = blockIdx.x * BN;
  const bool mValid  = (rowBase < M);

  const int lhalf = lane & 15;
  const int koff  = (lane < 16) ? 0 : 8;   // ISA 16-bit operand layout
  int aRowIdx = rowBase + lhalf;
  if (aRowIdx >= M) aRowIdx = M - 1;       // clamp (keeps all waves in barriers)
  const bf16_t* aRow = A + (long long)aRowIdx * lda + koff;

  const bf16_t* bTile0 = B + (long long)colBlock * ldb;   // tile origin, row-major [BN, K]
  const unsigned ldsBase = (unsigned)(uintptr_t)(&sB[0][0]);

  v8f acc[4] = {};
  const int KT = K / BK;

#if USE_TDM
  if (wave == 0) tdm_issue_tile(bTile0, ldsBase, ldb);
#endif

  for (int kt = 0; kt < KT; ++kt) {
    bf16_t* sBuf = &sB[kt & 1][0];
#if USE_TDM
    if (wave == 0) {
      if (kt + 1 < KT) {
        tdm_issue_tile(bTile0 + (long long)(kt + 1) * BK, ldsBase + ((kt + 1) & 1) * (BN * BK * 2), ldb);
        __builtin_amdgcn_s_wait_tensorcnt(1);
      } else {
        __builtin_amdgcn_s_wait_tensorcnt(0);
      }
    }
    __syncthreads();
#else
    // fallback: cooperative sync staging, single logical buffer per iteration
    __syncthreads();
#pragma unroll
    for (int i = 0; i < 4; ++i) {
      int ch = tid + i * 256;              // BN*BK*2 / 16 = 1024 16B chunks
      int r = ch >> 3, c = ch & 7;         // 8 chunks per 128B row
      *(u32x4*)&sBuf[r * BK + c * 8] =
          *(const u32x4*)(bTile0 + (long long)r * ldb + (long long)kt * BK + c * 8);
    }
    __syncthreads();
#endif

    const bf16_t* aTile = aRow + (long long)kt * BK;
    if (kt + 1 < KT) __builtin_prefetch(aTile + BK, 0, 1);

#pragma unroll
    for (int kk = 0; kk < BK; kk += 32) {
      v8bf a0 = *(const v8bf*)(aTile + kk);
      v8bf a1 = *(const v8bf*)(aTile + kk + 16);
      v16bf af;
#pragma unroll
      for (int i = 0; i < 8; ++i) { af[i] = a0[i]; af[i + 8] = a1[i]; }
      // Gather ALL four B fragments first so the ds_load_b128s issue as one
      // clause and the four WMMAs can run back-to-back (independent D regs).
      v16bf bfr[4];
#pragma unroll
      for (int nt = 0; nt < 4; ++nt) {
        const int rrow = waveNg * 64 + nt * 16 + lhalf;
        const bf16_t* bp = sBuf + rrow * BK + kk + koff;
        v8bf b0 = *(const v8bf*)(bp);
        v8bf b1 = *(const v8bf*)(bp + 16);
#pragma unroll
        for (int i = 0; i < 8; ++i) { bfr[nt][i] = b0[i]; bfr[nt][i + 8] = b1[i]; }
      }
#pragma unroll
      for (int nt = 0; nt < 4; ++nt) {
        acc[nt] = __builtin_amdgcn_wmma_f32_16x16x32_bf16(false, af, false, bfr[nt],
                                                          (short)0, acc[nt], false, false);
      }
    }
#if USE_TDM
    __syncthreads();   // protect buffer (kt&1) before it is re-issued at kt+1
#endif
  }

  if (!mValid) return;
  // C/D layout: lane<16 -> M=rowBase+r, N=lane; lane>=16 -> M=rowBase+8+r, N=lane-16
  const int mBase = rowBase + ((lane < 16) ? 0 : 8);
  float* cfB = Cf ? (Cf + (long long)bId * strideCf) : nullptr;
  bf16_t* cbB = Cb ? (Cb + (long long)bId * strideCb) : nullptr;
#pragma unroll
  for (int nt = 0; nt < 4; ++nt) {
    const int n = colBlock + waveNg * 64 + nt * 16 + lhalf;
    const float bv = bias ? bias[n] : 0.0f;
#pragma unroll
    for (int r = 0; r < 8; ++r) {
      float v = (acc[nt][r] + bv) * scale;
      if (act == 1) v = tanhf(v);
      long long m = mBase + r;
      if (cfB) cfB[m * (long long)ldcf + n] = v;
      if (cbB) cbB[m * (long long)ldcb + n] = (bf16_t)v;
    }
  }
}

// ---------------- GRU gate combine (one step) ----------------
__global__ void k_gru_gate(const float* __restrict__ x_proj, const float* __restrict__ hp,
                           float* __restrict__ h, bf16_t* __restrict__ h_b,
                           float* __restrict__ hs, bf16_t* __restrict__ hs_b,
                           bf16_t* __restrict__ cat_b, bf16_t* __restrict__ hsT_b, int t) {
  int idx = blockIdx.x * blockDim.x + threadIdx.x;   // 0 .. B*H-1
  int b = idx >> 10;
  int j = idx & 1023;
  const float* xp = x_proj + ((long long)(b * SS + t)) * (3 * HH);
  const float* hr_ = hp + (long long)b * (3 * HH);
  float xr = xp[j], xz = xp[HH + j], xn = xp[2 * HH + j];
  float hr = hr_[j], hz = hr_[HH + j], hn = hr_[2 * HH + j];
  float r = 1.0f / (1.0f + __expf(-(xr + hr)));
  float z = 1.0f / (1.0f + __expf(-(xz + hz)));
  float nn = tanhf(xn + r * hn);
  float hOld = h[idx];
  float hN = (1.0f - z) * nn + z * hOld;
  h[idx] = hN;
  h_b[idx] = (bf16_t)hN;                                     // rows 4..15 stay zero
  long long o = ((long long)(b * SS + t)) * HH + j;
  hs[o] = hN;
  hs_b[o] = (bf16_t)hN;
  cat_b[((long long)(b * SS + t)) * (2 * HH) + j] = (bf16_t)hN;   // left half of concat
  hsT_b[((long long)b * HH + j) * SS + t] = (bf16_t)hN;           // transposed copy for context GEMM
}

// ---------------- causal softmax row kernel ----------------
__global__ void k_softmax_causal(const float* __restrict__ scores, bf16_t* __restrict__ attn) {
  const int row = blockIdx.x;           // 0 .. B*S-1
  const int t = row & (SS - 1);
  const float* src = scores + (long long)row * SS;
  bf16_t* dst = attn + (long long)row * SS;
  const int tid = threadIdx.x, lane = tid & 31, wv = tid >> 5;
  __shared__ float red[8];

  float mx = -INFINITY;
  for (int s = tid; s <= t; s += 256) mx = fmaxf(mx, src[s]);
  for (int off = 16; off; off >>= 1) mx = fmaxf(mx, __shfl_xor(mx, off));
  if (lane == 0) red[wv] = mx;
  __syncthreads();
  if (wv == 0) {
    float v = (lane < 8) ? red[lane] : -INFINITY;
    for (int off = 4; off; off >>= 1) v = fmaxf(v, __shfl_xor(v, off));
    if (lane == 0) red[0] = v;
  }
  __syncthreads();
  mx = red[0];
  __syncthreads();

  float sum = 0.0f;
  for (int s = tid; s <= t; s += 256) sum += __expf(src[s] - mx);
  for (int off = 16; off; off >>= 1) sum += __shfl_xor(sum, off);
  if (lane == 0) red[wv] = sum;
  __syncthreads();
  if (wv == 0) {
    float v = (lane < 8) ? red[lane] : 0.0f;
    for (int off = 4; off; off >>= 1) v += __shfl_xor(v, off);
    if (lane == 0) red[0] = v;
  }
  __syncthreads();
  const float inv = 1.0f / red[0];

  for (int s = tid; s < SS; s += 256)
    dst[s] = (s <= t) ? (bf16_t)(__expf(src[s] - mx) * inv) : (bf16_t)0.0f;
}

// ---------------- host side ----------------
static inline dim3 gemm_grid(int M, int N, int batch) {
  return dim3((N + BN - 1) / BN, (M + BM - 1) / BM, batch);
}

extern "C" void kernel_launch(void* const* d_in, const int* in_sizes, int n_in,
                              void* d_out, int out_size, void* d_ws, size_t ws_size,
                              hipStream_t stream) {
  const int*   tokens = (const int*)  d_in[0];
  const float* emb    = (const float*)d_in[1];
  const float* w_ih   = (const float*)d_in[2];
  const float* w_hh   = (const float*)d_in[3];
  const float* b_ih   = (const float*)d_in[4];
  const float* b_hh   = (const float*)d_in[5];
  const float* wa     = (const float*)d_in[6];
  const float* ba     = (const float*)d_in[7];
  const float* wc     = (const float*)d_in[8];
  const float* bc     = (const float*)d_in[9];
  const float* wf     = (const float*)d_in[10];
  const float* bf     = (const float*)d_in[11];
  float* logits = (float*)d_out;

  // ---- workspace layout (256B aligned) ----
  char* ws = (char*)d_ws;
  size_t off = 0;
  auto alloc = [&](size_t bytes) { void* p = ws + off; off = (off + bytes + 255) & ~(size_t)255; return p; };
  bf16_t* w_ih_b = (bf16_t*)alloc((size_t)3 * HH * EE * 2);
  bf16_t* w_hh_b = (bf16_t*)alloc((size_t)3 * HH * HH * 2);
  bf16_t* wa_b   = (bf16_t*)alloc((size_t)HH * HH * 2);
  bf16_t* wc_b   = (bf16_t*)alloc((size_t)HH * 2 * HH * 2);
  bf16_t* wf_b   = (bf16_t*)alloc((size_t)VV * HH * 2);
  bf16_t* xb     = (bf16_t*)alloc((size_t)BB * SS * EE * 2);
  float*  x_proj = (float*) alloc((size_t)BB * SS * 3 * HH * 4);
  float*  hsta   = (float*) alloc((size_t)BB * HH * 4);
  bf16_t* h_b    = (bf16_t*)alloc((size_t)16 * HH * 2);
  float*  hp     = (float*) alloc((size_t)16 * 3 * HH * 4);
  float*  hs     = (float*) alloc((size_t)BB * SS * HH * 4);
  bf16_t* hs_b   = (bf16_t*)alloc((size_t)BB * SS * HH * 2);
  bf16_t* keys_b = (bf16_t*)alloc((size_t)BB * SS * HH * 2);
  float*  scores = (float*) alloc((size_t)BB * SS * SS * 4);
  bf16_t* attn_b = (bf16_t*)alloc((size_t)BB * SS * SS * 2);
  bf16_t* hsT_b  = (bf16_t*)alloc((size_t)BB * HH * SS * 2);
  bf16_t* cat_b  = (bf16_t*)alloc((size_t)BB * SS * 2 * HH * 2);
  bf16_t* comb_b = (bf16_t*)alloc((size_t)BB * SS * HH * 2);
  (void)ws_size; (void)in_sizes; (void)n_in; (void)out_size;

  auto cvt = [&](const float* s, bf16_t* d, long long n) {
    k_f32_to_bf16<<<(unsigned)((n + 255) / 256), 256, 0, stream>>>(s, d, n);
  };

  // 1) weight conversions
  cvt(w_ih, w_ih_b, (long long)3 * HH * EE);
  cvt(w_hh, w_hh_b, (long long)3 * HH * HH);
  cvt(wa,   wa_b,   (long long)HH * HH);
  cvt(wc,   wc_b,   (long long)HH * 2 * HH);
  cvt(wf,   wf_b,   (long long)VV * HH);

  // 2) embedding gather
  {
    long long n = (long long)BB * SS * EE;
    k_embed<<<(unsigned)((n + 255) / 256), 256, 0, stream>>>(tokens, emb, xb, n);
  }

  // 3) x_proj = x @ w_ih^T + b_ih   [4096,3072] f32
  k_wmma_gemm_bt<<<gemm_grid(BB * SS, 3 * HH, 1), 256, 0, stream>>>(
      xb, w_ih_b, b_ih, x_proj, nullptr,
      BB * SS, 3 * HH, EE, EE, EE, 3 * HH, 0, 0, 0, 0, 0, 1.0f, 0);

  // 4) GRU scan
  k_init_state<<<(16 * HH + 255) / 256, 256, 0, stream>>>(hsta, h_b);
  for (int t = 0; t < SS; ++t) {
    k_wmma_gemm_bt<<<gemm_grid(16, 3 * HH, 1), 256, 0, stream>>>(
        h_b, w_hh_b, b_hh, hp, nullptr,
        16, 3 * HH, HH, HH, HH, 3 * HH, 0, 0, 0, 0, 0, 1.0f, 0);
    k_gru_gate<<<(BB * HH) / 256, 256, 0, stream>>>(
        x_proj, hp, hsta, h_b, hs, hs_b, cat_b, hsT_b, t);
  }

  // 5) keys = hs @ wa^T + ba  (bf16 only)
  k_wmma_gemm_bt<<<gemm_grid(BB * SS, HH, 1), 256, 0, stream>>>(
      hs_b, wa_b, ba, nullptr, keys_b,
      BB * SS, HH, HH, HH, HH, 0, HH, 0, 0, 0, 0, 1.0f, 0);

  // 6) scores[b] = hs[b] @ keys[b]^T   (batched, f32)
  k_wmma_gemm_bt<<<gemm_grid(SS, SS, BB), 256, 0, stream>>>(
      hs_b, keys_b, nullptr, scores, nullptr,
      SS, SS, HH, HH, HH, SS, 0,
      (long long)SS * HH, (long long)SS * HH, (long long)SS * SS, 0, 1.0f, 0);

  // 7) causal softmax -> bf16 attn
  k_softmax_causal<<<BB * SS, 256, 0, stream>>>(scores, attn_b);

  // 8) context[b] = attn[b] @ hs[b]  == attn @ (hsT)^T ; bf16 into right half of concat
  k_wmma_gemm_bt<<<gemm_grid(SS, HH, BB), 256, 0, stream>>>(
      attn_b, hsT_b, nullptr, nullptr, cat_b + HH,
      SS, HH, SS, SS, SS, 0, 2 * HH,
      (long long)SS * SS, (long long)HH * SS, 0, (long long)SS * 2 * HH, 1.0f, 0);

  // 9) combined = tanh(cat @ wc^T + bc)  (bf16 only)
  k_wmma_gemm_bt<<<gemm_grid(BB * SS, HH, 1), 256, 0, stream>>>(
      cat_b, wc_b, bc, nullptr, comb_b,
      BB * SS, HH, 2 * HH, 2 * HH, 2 * HH, 0, HH, 0, 0, 0, 0, 1.0f, 1);

  // 10) logits = (combined @ wf^T + bf) / T  -> d_out f32
  k_wmma_gemm_bt<<<gemm_grid(BB * SS, VV, 1), 256, 0, stream>>>(
      comb_b, wf_b, bf, logits, nullptr,
      BB * SS, VV, HH, HH, HH, VV, 0, 0, 0, 0, 0, 1.25f, 0);
}
